// TinyAttn_51857435132299
// MI455X (gfx1250) — matrix operand order repaired and checked by reading, not compiled, not run on hardware
//
#include <hip/hip_runtime.h>
#include <stdint.h>

// ---------------------------------------------------------------------------
// CDNA5 (gfx1250, wave32) fused TinyAttn:
//   K1: qkv = x @ W1 + b1   (bf16 WMMA, q pre-scaled by 1/sqrt(64), qkv -> bf16 ws)
//   K2: flash attention     (bf16 WMMA scores + PV, online softmax, TDM-staged K)
//   K3: out = attn @ W2 + b2 (bf16 WMMA, f32 output)
// ---------------------------------------------------------------------------

typedef __attribute__((ext_vector_type(16))) __bf16 v16bf;
typedef __attribute__((ext_vector_type(8)))  float  v8f;
typedef __attribute__((ext_vector_type(4)))  unsigned uint32x4;
typedef __attribute__((ext_vector_type(8)))  unsigned uint32x8;

union Frag {
  v16bf bf;
  uint4 u4[2];
};

static __device__ __forceinline__ unsigned short f2bf(float f) {
  unsigned u = __float_as_uint(f);
  u += 0x7fffu + ((u >> 16) & 1u);   // round-to-nearest-even
  return (unsigned short)(u >> 16);
}

static __device__ __forceinline__ v8f wmma_bf16(const Frag& a, const Frag& b, v8f c) {
  // (neg_a, A, neg_b, B, c_mod, C, reuse_a, reuse_b)
  return __builtin_amdgcn_wmma_f32_16x16x32_bf16(false, a.bf, false, b.bf,
                                                 (short)0, c, false, false);
}

#define NB   2304        // 48*48 tokens per batch
#define DH   256         // hidden
#define QKVB (8LL * NB * DH)   // elements per q/k/v plane in ws

// ---------------------------------------------------------------------------
// Generic bf16-WMMA GEMM: C[M,N] = A[M,K](f32) * W[K,N](f32) + bias
// Block: 256 thr = 8 waves, tile 128x64 (wave tile 32x32), K-step 32.
// MODE 0: f32 out (N==256).  MODE 1: scatter bf16 into q/k/v ws, q *= 0.125.
// ---------------------------------------------------------------------------
template <int MODE>
__global__ __launch_bounds__(256) void gemm_bf16_kernel(
    const float* __restrict__ A, const float* __restrict__ W,
    const float* __restrict__ bias, unsigned short* __restrict__ qkv,
    float* __restrict__ outp, int K, int N) {
  __shared__ __align__(16) unsigned short sA[128 * 40];  // [128][32+8]
  __shared__ __align__(16) unsigned short sB[64 * 40];   // Bt: [64][32+8]

  const int tid  = threadIdx.x;
  const int lane = tid & 31;
  const int wave = tid >> 5;
  const int nlo  = lane & 15;
  const int hi   = lane >> 4;            // 0 / 1 (lane half)
  const int m0   = (wave & 3) * 32;
  const int n0   = (wave >> 2) * 32;
  const long long rowblk = (long long)blockIdx.y * 128;
  const int colblk = blockIdx.x * 64;

  const v8f zero8 = {0.f, 0.f, 0.f, 0.f, 0.f, 0.f, 0.f, 0.f};
  v8f acc[2][2];
  for (int i = 0; i < 2; ++i)
    for (int j = 0; j < 2; ++j) acc[i][j] = zero8;

  for (int k0 = 0; k0 < K; k0 += 32) {
    // stage A tile 128x32 (f32 -> bf16)
    for (int i = 0; i < 16; ++i) {
      int idx = tid + 256 * i;           // 4096 elements
      int r = idx >> 5, c = idx & 31;
      sA[r * 40 + c] = f2bf(A[(rowblk + r) * (long long)K + k0 + c]);
    }
    // stage B tile 32x64 transposed -> sB[n][k]
    for (int i = 0; i < 8; ++i) {
      int idx = tid + 256 * i;           // 2048 elements
      int kk = idx >> 6, n = idx & 63;
      sB[n * 40 + kk] = f2bf(W[(long long)(k0 + kk) * N + colblk + n]);
    }
    if (k0 + 32 < K)
      __builtin_prefetch(&A[(rowblk + (tid >> 3)) * (long long)K + k0 + 32], 0, 1);
    __syncthreads();

    Frag a0, a1, b0, b1;
    {
      int ma = m0 + nlo;
      a0.u4[0] = *(const uint4*)(sA + ma * 40 + hi * 16);
      a0.u4[1] = *(const uint4*)(sA + ma * 40 + hi * 16 + 8);
      int mb = m0 + 16 + nlo;
      a1.u4[0] = *(const uint4*)(sA + mb * 40 + hi * 16);
      a1.u4[1] = *(const uint4*)(sA + mb * 40 + hi * 16 + 8);
      int na = n0 + nlo;
      b0.u4[0] = *(const uint4*)(sB + na * 40 + hi * 16);
      b0.u4[1] = *(const uint4*)(sB + na * 40 + hi * 16 + 8);
      int nb = n0 + 16 + nlo;
      b1.u4[0] = *(const uint4*)(sB + nb * 40 + hi * 16);
      b1.u4[1] = *(const uint4*)(sB + nb * 40 + hi * 16 + 8);
    }
    acc[0][0] = wmma_bf16(a0, b0, acc[0][0]);
    acc[0][1] = wmma_bf16(a0, b1, acc[0][1]);
    acc[1][0] = wmma_bf16(a1, b0, acc[1][0]);
    acc[1][1] = wmma_bf16(a1, b1, acc[1][1]);
    __syncthreads();
  }

  // epilogue (C/D layout: lane col = nlo, rows r + hi*8)
  for (int i = 0; i < 2; ++i) {
    for (int j = 0; j < 2; ++j) {
      int col = colblk + n0 + j * 16 + nlo;
      float bv = bias[col];
      for (int r = 0; r < 8; ++r) {
        long long row = rowblk + m0 + i * 16 + r + hi * 8;
        float val = acc[i][j][r] + bv;
        if (MODE == 0) {
          outp[row * 256 + col] = val;
        } else {
          int sel = col >> 8, d = col & 255;
          if (sel == 0) val *= 0.125f;   // fold 1/sqrt(64) into q
          long long batch = row / NB, nr = row % NB;
          qkv[(((long long)sel * 8 + batch) * NB + nr) * DH + d] = f2bf(val);
        }
      }
    }
  }
}

// ---------------------------------------------------------------------------
// Flash attention: 4 waves (128 thr) per block, 64 q-rows per block,
// 16 q-rows per wave, 32 kv per step, d=256 accumulated in registers.
// Q tile lives in LDS (keeps VGPR count < 256); K tile staged by the
// Tensor Data Mover (tensor_load_to_lds + s_wait_tensorcnt); V transposed
// manually (TDM cannot transpose).
// Dynamic LDS layout (bytes):
//   sQ [64][264] @ 0       : 33792
//   sK [32][264] @ 33792   : 16896   (264 = 256 + 8 pad -> TDM pad 4 dw / 128 dw)
//   sV [256][40] @ 50688   : 20480
//   sP [4][16][40] @ 71168 :  5120   total 76288
// ---------------------------------------------------------------------------
#define FLASH_LDS_BYTES 76288

__global__ __launch_bounds__(128, 1) void flash_attn_kernel(
    const unsigned short* __restrict__ qw, const unsigned short* __restrict__ kw,
    const unsigned short* __restrict__ vw, float* __restrict__ o) {
  extern __shared__ __align__(16) char smem[];
  unsigned short* sQ = (unsigned short*)(smem);
  unsigned short* sK = (unsigned short*)(smem + 33792);
  unsigned short* sV = (unsigned short*)(smem + 50688);
  unsigned short* sP = (unsigned short*)(smem + 71168);

  const int tid  = threadIdx.x;
  const int lane = tid & 31;
  const int wave = tid >> 5;
  const int nlo  = lane & 15;
  const int hi   = lane >> 4;
  const int batch = blockIdx.y;
  const int rowblk = blockIdx.x * 64;
  const size_t bbase = (size_t)batch * NB * DH;

  // stage Q tile 64x256 -> sQ[row][d] (row-major == A-fragment layout)
  {
    const unsigned short* qs = qw + bbase + (size_t)rowblk * DH;
    for (int i = 0; i < 16; ++i) {
      int idx = tid + 128 * i;           // 2048 x uint4
      int r = idx >> 5, c = idx & 31;
      *(uint4*)(sQ + r * 264 + c * 8) = *(const uint4*)(qs + r * DH + c * 8);
    }
  }
  __syncthreads();

  const v8f zero8 = {0.f, 0.f, 0.f, 0.f, 0.f, 0.f, 0.f, 0.f};
  v8f oacc[16];
  for (int t = 0; t < 16; ++t) oacc[t] = zero8;
  float mrow[8], lrow[8];
  for (int r = 0; r < 8; ++r) { mrow[r] = -3.0e38f; lrow[r] = 0.f; }

  unsigned short* pw = sP + wave * 16 * 40;
  const unsigned ldsK = (unsigned)(uintptr_t)sK;   // LDS byte offset of K tile

  for (int kv0 = 0; kv0 < NB; kv0 += 32) {
    // ---- stage K rows [kv0, kv0+32) via Tensor Data Mover (wave 0 issues)
    if (wave == 0) {
      unsigned long long ga =
          (unsigned long long)(uintptr_t)(kw + bbase + (size_t)kv0 * DH);
      uint32x4 g0;
      g0[0] = 1u;                                   // count=1 user descriptor
      g0[1] = ldsK;                                 // lds_addr
      g0[2] = (unsigned)(ga & 0xffffffffu);         // global_addr[31:0]
      g0[3] = (unsigned)((ga >> 32) & 0x01ffffffu)  // global_addr[56:32]
              | (2u << 30);                         // type = 2 (image)
      uint32x8 g1;
      g1[0] = (1u << 16)        // data_size = 2 bytes
            | (1u << 20)        // pad_enable
            | (6u << 22)        // pad_interval: 128 DWORDs (one 512B row)
            | (3u << 25);       // pad_amount: 4 DWORDs (16B -> 264-short rows)
      g1[1] = (256u & 0xffffu) << 16;  // tensor_dim0 = 256 (lo16)
      g1[2] = (32u << 16);             // dim0 hi16=0 | tensor_dim1 = 32 (lo16)
      g1[3] = (256u << 16);            // dim1 hi16=0 | tile_dim0 = 256
      g1[4] = 32u;                     // tile_dim1 = 32, tile_dim2 = 0
      g1[5] = 256u;                    // tensor_dim0_stride = 256 (lo32)
      g1[6] = 0u;                      // stride hi16 = 0, dim1_stride lo = 0
      g1[7] = 0u;
      uint32x4 gz = {0u, 0u, 0u, 0u};  // groups 2/3 unused (2D tensor)
      asm volatile("tensor_load_to_lds %0, %1, %2, %3"
                   :
                   : "s"(g0), "s"(g1), "s"(gz), "s"(gz)
                   : "memory");
    }
    // ---- stage V transposed: sV[d][kv] (manual; TDM cannot transpose)
    const unsigned short* vs = vw + bbase + (size_t)kv0 * DH;
    for (int i = 0; i < 32; ++i) {
      int idx = tid + 128 * i;           // 4096 x uint (pairs of bf16 along d)
      int kv = idx >> 7, c = idx & 127;
      unsigned u = *(const unsigned*)(vs + kv * DH + c * 2);
      sV[(c * 2 + 0) * 40 + kv] = (unsigned short)(u & 0xffffu);
      sV[(c * 2 + 1) * 40 + kv] = (unsigned short)(u >> 16);
    }
    if (kv0 + 32 < NB)
      __builtin_prefetch(vw + bbase + (size_t)(kv0 + 32) * DH + (tid << 4), 0, 1);
    if (wave == 0) __builtin_amdgcn_s_wait_tensorcnt(0);
    __syncthreads();

    // ---- scores: S0 (kv 0..15), S1 (kv 16..31), K-dim = d = 256
    v8f s0 = zero8, s1 = zero8;
    const int mq = wave * 16 + nlo;
    for (int kk = 0; kk < 8; ++kk) {
      Frag aq, bk0, bk1;
      aq.u4[0]  = *(const uint4*)(sQ + mq * 264 + kk * 32 + hi * 16);
      aq.u4[1]  = *(const uint4*)(sQ + mq * 264 + kk * 32 + hi * 16 + 8);
      bk0.u4[0] = *(const uint4*)(sK + nlo * 264 + kk * 32 + hi * 16);
      bk0.u4[1] = *(const uint4*)(sK + nlo * 264 + kk * 32 + hi * 16 + 8);
      bk1.u4[0] = *(const uint4*)(sK + (nlo + 16) * 264 + kk * 32 + hi * 16);
      bk1.u4[1] = *(const uint4*)(sK + (nlo + 16) * 264 + kk * 32 + hi * 16 + 8);
      s0 = wmma_bf16(aq, bk0, s0);
      s1 = wmma_bf16(aq, bk1, s1);
    }

    // ---- online softmax (xor-shuffle row reductions within 16-lane halves)
    float alpha[8];
    for (int r = 0; r < 8; ++r) {
      float mx = fmaxf(s0[r], s1[r]);
      mx = fmaxf(mx, __shfl_xor(mx, 1));
      mx = fmaxf(mx, __shfl_xor(mx, 2));
      mx = fmaxf(mx, __shfl_xor(mx, 4));
      mx = fmaxf(mx, __shfl_xor(mx, 8));
      float mnew = fmaxf(mrow[r], mx);
      alpha[r] = __expf(mrow[r] - mnew);
      mrow[r] = mnew;
      float p0 = __expf(s0[r] - mnew);
      float p1 = __expf(s1[r] - mnew);
      s0[r] = p0;
      s1[r] = p1;
      float rs = p0 + p1;
      rs += __shfl_xor(rs, 1);
      rs += __shfl_xor(rs, 2);
      rs += __shfl_xor(rs, 4);
      rs += __shfl_xor(rs, 8);
      lrow[r] = lrow[r] * alpha[r] + rs;
    }
    for (int t = 0; t < 16; ++t)
      for (int r = 0; r < 8; ++r) oacc[t][r] *= alpha[r];

    // ---- P (f32 C-layout) -> LDS bf16 -> A-fragment (wave-private buffer)
    for (int r = 0; r < 8; ++r) {
      int m = r + hi * 8;
      pw[m * 40 + nlo]      = f2bf(s0[r]);
      pw[m * 40 + 16 + nlo] = f2bf(s1[r]);
    }
    Frag aP;
    aP.u4[0] = *(const uint4*)(pw + nlo * 40 + hi * 16);
    aP.u4[1] = *(const uint4*)(pw + nlo * 40 + hi * 16 + 8);

    // ---- O += P(16x32) @ V(32x256), 16 d-chunks of 16
    for (int nd = 0; nd < 16; ++nd) {
      Frag bv;
      int d = nd * 16 + nlo;
      bv.u4[0] = *(const uint4*)(sV + d * 40 + hi * 16);
      bv.u4[1] = *(const uint4*)(sV + d * 40 + hi * 16 + 8);
      oacc[nd] = wmma_bf16(aP, bv, oacc[nd]);
    }
    __syncthreads();
  }

  // ---- normalize and store f32
  for (int r = 0; r < 8; ++r) lrow[r] = 1.0f / lrow[r];
  float* dst = o + bbase + (size_t)rowblk * DH;
  for (int nd = 0; nd < 16; ++nd) {
    int col = nd * 16 + nlo;
    for (int r = 0; r < 8; ++r) {
      int m = wave * 16 + r + hi * 8;
      dst[(size_t)m * DH + col] = oacc[nd][r] * lrow[r];
    }
  }
}

// ---------------------------------------------------------------------------
extern "C" void kernel_launch(void* const* d_in, const int* in_sizes, int n_in,
                              void* d_out, int out_size, void* d_ws, size_t ws_size,
                              hipStream_t stream) {
  (void)in_sizes; (void)n_in; (void)out_size; (void)ws_size;
  const float* x  = (const float*)d_in[0];   // (8,48,48,512)
  const float* W1 = (const float*)d_in[1];   // (512,768)
  const float* b1 = (const float*)d_in[2];   // (768)
  const float* W2 = (const float*)d_in[3];   // (256,256)
  const float* b2 = (const float*)d_in[4];   // (256)
  float* out = (float*)d_out;                // (8,48,48,256)

  unsigned short* qkv = (unsigned short*)d_ws;           // 3 * 8*2304*256 bf16
  float* attn = (float*)((char*)d_ws + (size_t)(3 * QKVB) * sizeof(unsigned short));

  const int M = 8 * NB;  // 18432

  // K1: qkv = x @ W1 + b1 -> scattered bf16 q/k/v planes
  dim3 g1(768 / 64, M / 128);
  gemm_bf16_kernel<1><<<g1, 256, 0, stream>>>(x, W1, b1, qkv, nullptr, 512, 768);

  // K2: flash attention per (row-block, batch)
  dim3 g2(NB / 64, 8);
  flash_attn_kernel<<<g2, 128, FLASH_LDS_BYTES, stream>>>(qkv, qkv + QKVB,
                                                          qkv + 2 * QKVB, attn);

  // K3: out = attn @ W2 + b2
  dim3 g3(256 / 64, M / 128);
  gemm_bf16_kernel<0><<<g3, 256, 0, stream>>>(attn, W2, b2, nullptr, out, 256, 256);
}
